// ViT_18219251270124
// MI455X (gfx1250) — matrix operand order repaired and checked by reading, not compiled, not run on hardware
//
#include <hip/hip_runtime.h>
#include <hip/hip_bf16.h>
#include <math.h>

typedef _Float16 half_t;
typedef __attribute__((ext_vector_type(16))) _Float16 v16h;
typedef __attribute__((ext_vector_type(8)))  _Float16 v8h;
typedef __attribute__((ext_vector_type(8)))  float    v8f;
typedef __attribute__((vector_size(16)))     int      i32x4;

union V16U { v16h v; v8h h[2]; };

#define L_LAYERS 12
#define DIM      768
#define FFDIM    3072
#define NHEADS   12
#define DHEAD    64
#define BATCH    32
#define NPATCH   576      // 24*24
#define SPAD     592      // 588 padded up to multiple of 16
#define SVALID   588      // reference sequence length after its own padding
#define NCLS     1000
#define MQ       (BATCH * SPAD)    // 18944  (divisible by 128)
#define MP       (BATCH * NPATCH)  // 18432  (divisible by 128)

#define AS1 __attribute__((address_space(1)))
#define AS3 __attribute__((address_space(3)))

#if __has_builtin(__builtin_amdgcn_global_load_async_to_lds_b128)
#define USE_ASYNC_B128 1
#endif

// 16-byte global -> LDS copy (async when the CDNA5 builtin is available)
__device__ __forceinline__ void stage16(const half_t* __restrict__ g,
                                        half_t* __restrict__ l) {
#if defined(USE_ASYNC_B128)
  __builtin_amdgcn_global_load_async_to_lds_b128((AS1 i32x4*)g,
                                                 (AS3 i32x4*)l, 0, 0);
#else
  *(uint4*)l = *(const uint4*)g;
#endif
}

__device__ __forceinline__ void wait_async() {
#if defined(USE_ASYNC_B128)
#if __has_builtin(__builtin_amdgcn_s_wait_asynccnt)
  __builtin_amdgcn_s_wait_asynccnt(0);
#else
  asm volatile("s_wait_asynccnt 0" ::: "memory");
#endif
#endif
}

// ---------------------------------------------------------------------------
// weight convert + transpose: src (L,K,N) f32 -> dst (L,N,K) f16
// ---------------------------------------------------------------------------
__global__ void transpose_cvt_kernel(const float* __restrict__ src,
                                     half_t* __restrict__ dst,
                                     int Ld, int K, int N) {
  long idx = (long)blockIdx.x * blockDim.x + threadIdx.x;
  long per = (long)K * N;
  long total = (long)Ld * per;
  if (idx >= total) return;
  int l = (int)(idx / per);
  long rem = idx - (long)l * per;
  int kk = (int)(rem / N);
  int nn = (int)(rem - (long)kk * N);
  dst[(long)l * per + (long)nn * K + kk] = (half_t)src[idx];
}

// ---------------------------------------------------------------------------
// patchify: x (B,3,384,384) -> xp (B*576, 768) f16, col = (p*16+q)*3+c
// ---------------------------------------------------------------------------
__global__ void patchify_kernel(const float* __restrict__ x,
                                half_t* __restrict__ xp) {
  long idx = (long)blockIdx.x * blockDim.x + threadIdx.x;
  long total = (long)MP * DIM;
  if (idx >= total) return;
  int col = (int)(idx % DIM);
  long rowg = idx / DIM;
  int pr = (int)(rowg % NPATCH);
  int b  = (int)(rowg / NPATCH);
  int c = col % 3;
  int q = (col / 3) & 15;
  int p = col / 48;
  int hp = pr / 24, wp = pr % 24;
  long xi = (((long)b * 3 + c) * 384 + (hp * 16 + p)) * 384 + (wp * 16 + q);
  xp[idx] = (half_t)x[xi];
}

// ---------------------------------------------------------------------------
// carry setup: carry (B, SPAD, 768) f32
// ---------------------------------------------------------------------------
__global__ void setup_carry_kernel(const float* __restrict__ pe32,
                                   const float* __restrict__ cls,
                                   const float* __restrict__ pos,
                                   float* __restrict__ carry) {
  long idx = (long)blockIdx.x * blockDim.x + threadIdx.x;
  long total = (long)MQ * DIM;
  if (idx >= total) return;
  int d = (int)(idx % DIM);
  long rowg = idx / DIM;
  int s = (int)(rowg % SPAD);
  int b = (int)(rowg / SPAD);
  float v = 0.0f;
  if (s == 0)           v = cls[d] + pos[d];
  else if (s <= NPATCH) v = pe32[((long)b * NPATCH + (s - 1)) * DIM + d]
                           + pos[(long)s * DIM + d];
  carry[idx] = v;
}

// ---------------------------------------------------------------------------
// LayerNorm per row
// ---------------------------------------------------------------------------
__device__ __forceinline__ float block_reduce_sum(float v, float* red) {
  red[threadIdx.x] = v;
  __syncthreads();
  for (int s = 128; s > 0; s >>= 1) {
    if ((int)threadIdx.x < s) red[threadIdx.x] += red[threadIdx.x + s];
    __syncthreads();
  }
  float r = red[0];
  __syncthreads();
  return r;
}

__global__ __launch_bounds__(256) void ln_rows_kernel(
    const float* __restrict__ x, const float* __restrict__ g,
    const float* __restrict__ bt, half_t* __restrict__ out) {
  __shared__ float red[256];
  long row = blockIdx.x;
  const float* xr = x + row * DIM;
  float s = 0.0f;
  for (int d = threadIdx.x; d < DIM; d += 256) s += xr[d];
  float mu = block_reduce_sum(s, red) * (1.0f / DIM);
  float vs = 0.0f;
  for (int d = threadIdx.x; d < DIM; d += 256) {
    float t = xr[d] - mu;
    vs += t * t;
  }
  float var = block_reduce_sum(vs, red) * (1.0f / DIM);
  float rstd = rsqrtf(var + 1e-5f);
  for (int d = threadIdx.x; d < DIM; d += 256)
    out[row * DIM + d] = (half_t)((xr[d] - mu) * rstd * g[d] + bt[d]);
}

// ---------------------------------------------------------------------------
// WMMA GEMM: C = A(MxK,f16) @ Bt(NxK,f16 K-major) [+bias][gelu][residual f32]
// 256 threads (8 waves), macro tile 128x64, K step 32, async LDS staging.
// OP: 0 f16 out; 1 f16 out +bias +GELU; 2 f32 carry += (+bias); 3 f32 = +bias
// Requires M%128==0, N%64==0, K%32==0.
// ---------------------------------------------------------------------------
template <int OP>
__global__ __launch_bounds__(256) void gemm_wmma_kernel(
    const half_t* __restrict__ A, const half_t* __restrict__ Bt,
    const float* __restrict__ bias, float* __restrict__ Cf,
    half_t* __restrict__ Ch, int M, int N, int K) {
  __shared__ __align__(16) half_t As[128 * 32];
  __shared__ __align__(16) half_t Bs[64 * 32];   // N-major, K contiguous

  const int tid  = threadIdx.x;
  const int wave = tid >> 5;
  const int lane = tid & 31;
  const int rowBase = blockIdx.y * 128;
  const int colBase = blockIdx.x * 64;

  v8f acc[4] = {};

  const int aRow = tid >> 1;            // 0..127
  const int aSeg = (tid & 1) * 16;      // 0 / 16 halves
  const int bRow = tid >> 2;            // 0..63 (N index)
  const int bSeg = (tid & 3) * 8;       // 0..24 halves

  const int mrow = (wave << 4) + (lane & 15);   // wave-local A row
  const int kb   = (lane >> 4) * 16;            // A-frag K base (0/16)
  const int kr0  = (lane >> 4) * 16;            // B-frag K base (0/16)

  const half_t* aG = A + (long)(rowBase + aRow) * K + aSeg;
  const half_t* bG = Bt + (long)(colBase + bRow) * K + bSeg;

  for (int kk = 0; kk < K; kk += 32) {
    if (kk + 32 < K) {
      __builtin_prefetch(aG + kk + 32, 0, 0);
      __builtin_prefetch(bG + kk + 32, 0, 0);
    }
    // stage A tile 128x32 (2x16B per thread) and B tile 64x32 (1x16B)
    stage16(aG + kk,     As + aRow * 32 + aSeg);
    stage16(aG + kk + 8, As + aRow * 32 + aSeg + 8);
    stage16(bG + kk,     Bs + bRow * 32 + bSeg);
    wait_async();
    __syncthreads();

    V16U af;
    af.h[0] = *(const v8h*)(As + mrow * 32 + kb);
    af.h[1] = *(const v8h*)(As + mrow * 32 + kb + 8);
#pragma unroll
    for (int t = 0; t < 4; ++t) {
      const int ncol = (t << 4) + (lane & 15);
      V16U bf;
      bf.h[0] = *(const v8h*)(Bs + ncol * 32 + kr0);
      bf.h[1] = *(const v8h*)(Bs + ncol * 32 + kr0 + 8);
      acc[t] = __builtin_amdgcn_wmma_f32_16x16x32_f16(
          false, af.v, false, bf.v, (short)0, acc[t], false, false);
    }
    __syncthreads();
  }

  // epilogue: C-frag layout -> lane = col, VGPR r -> row r + 8*(lane>>4)
  const int lrow = (wave << 4) + ((lane >> 4) << 3);
  const int lcol = lane & 15;
#pragma unroll
  for (int t = 0; t < 4; ++t) {
    int col = colBase + (t << 4) + lcol;
    float bv = 0.0f;
    if (OP != 0 && bias != nullptr) bv = bias[col];
#pragma unroll
    for (int r = 0; r < 8; ++r) {
      long idx = (long)(rowBase + lrow + r) * N + col;
      float v = acc[t][r];
      if (OP == 0) {
        Ch[idx] = (half_t)v;
      } else if (OP == 1) {
        v += bv;
        v = 0.5f * v * (1.0f + erff(v * 0.70710678118654752f));
        Ch[idx] = (half_t)v;
      } else if (OP == 2) {
        Cf[idx] += v + bv;
      } else {
        Cf[idx] = v + bv;
      }
    }
  }
}

// ---------------------------------------------------------------------------
// Flash attention: one wave per (16-query tile, head, batch).
// q/k/v/o layout (B,SPAD,DIM) f16; head hh at cols [hh*64, hh*64+64).
// Scores scaled 1/768; key cols >= 588 masked to -inf.
// ---------------------------------------------------------------------------
__global__ __launch_bounds__(32) void flash_attn_kernel(
    const half_t* __restrict__ q, const half_t* __restrict__ k,
    const half_t* __restrict__ v, half_t* __restrict__ o) {
  const int lane = threadIdx.x;
  const int qt = blockIdx.x;   // 0..36
  const int hh = blockIdx.y;
  const int b  = blockIdx.z;
  const long baseRow = (long)b * SPAD;

  __shared__ __align__(16) half_t Ps[16 * 32];
  __shared__ __align__(16) half_t Vs[32 * 64];

  const int kb  = (lane >> 4) * 16;    // A-frag K base
  const int kd0 = (lane >> 4) * 16;    // B-frag K base

  // Q fragments (dh dims 0..31 / 32..63), contiguous 16B chunks
  const int qrow = qt * 16 + (lane & 15);
  const half_t* qf = q + (baseRow + qrow) * DIM + hh * DHEAD;
  V16U aq0, aq1;
  aq0.h[0] = *(const v8h*)(qf + kb);
  aq0.h[1] = *(const v8h*)(qf + kb + 8);
  aq1.h[0] = *(const v8h*)(qf + 32 + kb);
  aq1.h[1] = *(const v8h*)(qf + 32 + kb + 8);

  v8f oacc[4] = {};
  float mrow[8], ssum[8];
#pragma unroll
  for (int r = 0; r < 8; ++r) { mrow[r] = -3e38f; ssum[r] = 0.0f; }

  const float scale = 1.0f / (float)DIM;

  for (int p = 0; p < 19; ++p) {   // 19 pairs of 16-key tiles >= SPAD
    // stage V tile (32 keys x 64 dims) into LDS: lane owns one key row
    {
      const int vrow = p * 32 + lane;
      if (vrow < SPAD) {
        const half_t* vg = v + (baseRow + vrow) * DIM + hh * DHEAD;
        half_t* vl = Vs + lane * 64;
#pragma unroll
        for (int c = 0; c < 8; ++c) stage16(vg + c * 8, vl + c * 8);
      } else {
        v8h z = {};
#pragma unroll
        for (int c = 0; c < 8; ++c) *(v8h*)(Vs + lane * 64 + c * 8) = z;
      }
    }

    // scores: Q @ K^T (K fragment is contiguous along dh in memory)
    v8f sc[2];
#pragma unroll
    for (int jj = 0; jj < 2; ++jj) {
      const int krow = p * 32 + jj * 16 + (lane & 15);
      const bool inb = krow < SPAD;
      V16U bk0, bk1;
      if (inb) {
        const half_t* kg = k + (baseRow + krow) * DIM + hh * DHEAD;
        bk0.h[0] = *(const v8h*)(kg + kd0);
        bk0.h[1] = *(const v8h*)(kg + kd0 + 8);
        bk1.h[0] = *(const v8h*)(kg + 32 + kd0);
        bk1.h[1] = *(const v8h*)(kg + 32 + kd0 + 8);
      } else {
        v8h z = {};
        bk0.h[0] = z; bk0.h[1] = z; bk1.h[0] = z; bk1.h[1] = z;
      }
      v8f s = {};
      s = __builtin_amdgcn_wmma_f32_16x16x32_f16(false, aq0.v, false, bk0.v,
                                                 (short)0, s, false, false);
      s = __builtin_amdgcn_wmma_f32_16x16x32_f16(false, aq1.v, false, bk1.v,
                                                 (short)0, s, false, false);
      const bool valid = krow < SVALID;
#pragma unroll
      for (int r = 0; r < 8; ++r) s[r] = valid ? s[r] * scale : -3e38f;
      sc[jj] = s;
    }

    // online softmax (stats per (vgpr-row, 16-lane half))
    float alpha[8];
#pragma unroll
    for (int r = 0; r < 8; ++r) {
      float t = fmaxf(sc[0][r], sc[1][r]);
      t = fmaxf(t, __shfl_xor(t, 1, 32));
      t = fmaxf(t, __shfl_xor(t, 2, 32));
      t = fmaxf(t, __shfl_xor(t, 4, 32));
      t = fmaxf(t, __shfl_xor(t, 8, 32));
      float mn = fmaxf(mrow[r], t);
      alpha[r] = expf(mrow[r] - mn);
      mrow[r] = mn;
      sc[0][r] = expf(sc[0][r] - mn);
      sc[1][r] = expf(sc[1][r] - mn);
      float rs = sc[0][r] + sc[1][r];
      rs += __shfl_xor(rs, 1, 32);
      rs += __shfl_xor(rs, 2, 32);
      rs += __shfl_xor(rs, 4, 32);
      rs += __shfl_xor(rs, 8, 32);
      ssum[r] = ssum[r] * alpha[r] + rs;
    }
#pragma unroll
    for (int t4 = 0; t4 < 4; ++t4)
#pragma unroll
      for (int r = 0; r < 8; ++r) oacc[t4][r] *= alpha[r];

    // P (C-layout f32) -> LDS -> A-fragment f16 (transpose)
    {
      const int prow_off = (lane >> 4) << 3;
      const int pcol = lane & 15;
#pragma unroll
      for (int r = 0; r < 8; ++r) {
        Ps[(prow_off + r) * 32 + pcol]      = (half_t)sc[0][r];
        Ps[(prow_off + r) * 32 + 16 + pcol] = (half_t)sc[1][r];
      }
    }
    wait_async();          // V tile async copies complete
    __syncthreads();       // Ps + Vs visible wave-wide

    V16U pa;
    {
      const int prow = lane & 15;
      pa.h[0] = *(const v8h*)(Ps + prow * 32 + kb);
      pa.h[1] = *(const v8h*)(Ps + prow * 32 + kb + 8);
    }

    // O += P @ V (V fragment gathered from LDS, transposed)
#pragma unroll
    for (int t4 = 0; t4 < 4; ++t4) {
      const int dcol = t4 * 16 + (lane & 15);
      v16h bv;
#pragma unroll
      for (int h = 0; h < 16; ++h) bv[h] = Vs[(kd0 + h) * 64 + dcol];
      oacc[t4] = __builtin_amdgcn_wmma_f32_16x16x32_f16(
          false, pa.v, false, bv, (short)0, oacc[t4], false, false);
    }
    __syncthreads();       // before next-iteration LDS overwrite
  }

  // normalize + store
#pragma unroll
  for (int t4 = 0; t4 < 4; ++t4) {
    const int dcol = t4 * 16 + (lane & 15);
#pragma unroll
    for (int r = 0; r < 8; ++r) {
      int row = qt * 16 + ((lane >> 4) << 3) + r;
      o[(baseRow + row) * DIM + hh * DHEAD + dcol] =
          (half_t)(oacc[t4][r] / ssum[r]);
    }
  }
}

// ---------------------------------------------------------------------------
// head: logits[b,:] = LN(carry[b,0,:]) @ head_W + head_b
// ---------------------------------------------------------------------------
__global__ __launch_bounds__(256) void head_kernel(
    const float* __restrict__ carry, const float* __restrict__ g,
    const float* __restrict__ bt, const float* __restrict__ W,
    const float* __restrict__ hb, float* __restrict__ out) {
  __shared__ float row[DIM];
  __shared__ float red[256];
  const int b = blockIdx.x;
  const float* xr = carry + (long)b * SPAD * DIM;
  float s = 0.0f;
  for (int d = threadIdx.x; d < DIM; d += 256) s += xr[d];
  float mu = block_reduce_sum(s, red) * (1.0f / DIM);
  float vs = 0.0f;
  for (int d = threadIdx.x; d < DIM; d += 256) {
    float t = xr[d] - mu;
    vs += t * t;
  }
  float var = block_reduce_sum(vs, red) * (1.0f / DIM);
  float rstd = rsqrtf(var + 1e-5f);
  for (int d = threadIdx.x; d < DIM; d += 256)
    row[d] = (xr[d] - mu) * rstd * g[d] + bt[d];
  __syncthreads();
  for (int c = threadIdx.x; c < NCLS; c += 256) {
    float acc = hb[c];
    for (int d = 0; d < DIM; ++d) acc += row[d] * W[(long)d * NCLS + c];
    out[(long)b * NCLS + c] = acc;
  }
}

// ---------------------------------------------------------------------------
// host orchestration
// ---------------------------------------------------------------------------
static inline size_t alignUp256(size_t x) { return (x + 255) & ~(size_t)255; }

extern "C" void kernel_launch(void* const* d_in, const int* in_sizes, int n_in,
                              void* d_out, int out_size, void* d_ws,
                              size_t ws_size, hipStream_t stream) {
  (void)in_sizes; (void)n_in; (void)out_size; (void)ws_size;

  const float* x       = (const float*)d_in[0];
  const float* patch_W = (const float*)d_in[1];
  const float* patch_b = (const float*)d_in[2];
  const float* cls_tok = (const float*)d_in[3];
  const float* pos_emb = (const float*)d_in[4];
  const float* ln1_g   = (const float*)d_in[5];
  const float* ln1_b   = (const float*)d_in[6];
  const float* Wq      = (const float*)d_in[7];
  const float* Wk      = (const float*)d_in[8];
  const float* Wv      = (const float*)d_in[9];
  const float* Wo      = (const float*)d_in[10];
  const float* ln2_g   = (const float*)d_in[11];
  const float* ln2_b   = (const float*)d_in[12];
  const float* mlp_W1  = (const float*)d_in[13];
  const float* mlp_b1  = (const float*)d_in[14];
  const float* mlp_W2  = (const float*)d_in[15];
  const float* mlp_b2  = (const float*)d_in[16];
  const float* lnf_g   = (const float*)d_in[17];
  const float* lnf_b   = (const float*)d_in[18];
  const float* head_W  = (const float*)d_in[19];
  const float* head_b  = (const float*)d_in[20];

  char* ws = (char*)d_ws;
  size_t off = 0;
  auto take = [&](size_t bytes) -> char* {
    char* p = ws + off;
    off += alignUp256(bytes);
    return p;
  };
  const size_t WQK = (size_t)L_LAYERS * DIM * DIM;
  const size_t WFF = (size_t)L_LAYERS * DIM * FFDIM;

  half_t* wq16 = (half_t*)take(WQK * 2);   // transposed (L, N, K) f16
  half_t* wk16 = (half_t*)take(WQK * 2);
  half_t* wv16 = (half_t*)take(WQK * 2);
  half_t* wo16 = (half_t*)take(WQK * 2);
  half_t* w116 = (half_t*)take(WFF * 2);   // (L, FF, D)
  half_t* w216 = (half_t*)take(WFF * 2);   // (L, D, FF)
  half_t* pw16 = (half_t*)take((size_t)DIM * DIM * 2);
  half_t* xp16 = (half_t*)take((size_t)MP * DIM * 2);
  float*  pe32 = (float*) take((size_t)MP * DIM * 4);
  float*  carry= (float*) take((size_t)MQ * DIM * 4);
  half_t* hN16 = (half_t*)take((size_t)MQ * DIM * 2);
  half_t* q16  = (half_t*)take((size_t)MQ * DIM * 2);
  half_t* k16  = (half_t*)take((size_t)MQ * DIM * 2);
  half_t* v16  = (half_t*)take((size_t)MQ * DIM * 2);
  half_t* o16  = (half_t*)take((size_t)MQ * DIM * 2);
  half_t* hid16= (half_t*)take((size_t)MQ * FFDIM * 2);

  auto tcvt = [&](const float* s, half_t* d, int Ld, int K, int N) {
    long n = (long)Ld * K * N;
    transpose_cvt_kernel<<<(int)((n + 255) / 256), 256, 0, stream>>>(s, d, Ld,
                                                                     K, N);
  };
  tcvt(Wq, wq16, L_LAYERS, DIM, DIM);
  tcvt(Wk, wk16, L_LAYERS, DIM, DIM);
  tcvt(Wv, wv16, L_LAYERS, DIM, DIM);
  tcvt(Wo, wo16, L_LAYERS, DIM, DIM);
  tcvt(mlp_W1, w116, L_LAYERS, DIM, FFDIM);
  tcvt(mlp_W2, w216, L_LAYERS, FFDIM, DIM);
  tcvt(patch_W, pw16, 1, DIM, DIM);

  // ---- patch embed ----
  {
    long n = (long)MP * DIM;
    patchify_kernel<<<(int)((n + 255) / 256), 256, 0, stream>>>(x, xp16);
    dim3 grid(DIM / 64, MP / 128);
    gemm_wmma_kernel<3><<<grid, 256, 0, stream>>>(xp16, pw16, patch_b, pe32,
                                                  nullptr, MP, DIM, DIM);
    long nc = (long)MQ * DIM;
    setup_carry_kernel<<<(int)((nc + 255) / 256), 256, 0, stream>>>(
        pe32, cls_tok, pos_emb, carry);
  }

  // ---- encoder layers ----
  dim3 gProj(DIM / 64, MQ / 128);
  dim3 gFF(FFDIM / 64, MQ / 128);
  dim3 gAttn(SPAD / 16, NHEADS, BATCH);

  for (int l = 0; l < L_LAYERS; ++l) {
    ln_rows_kernel<<<MQ, 256, 0, stream>>>(carry, ln1_g + (long)l * DIM,
                                           ln1_b + (long)l * DIM, hN16);
    const long wOff = (long)l * DIM * DIM;
    gemm_wmma_kernel<0><<<gProj, 256, 0, stream>>>(hN16, wq16 + wOff, nullptr,
                                                   nullptr, q16, MQ, DIM, DIM);
    gemm_wmma_kernel<0><<<gProj, 256, 0, stream>>>(hN16, wk16 + wOff, nullptr,
                                                   nullptr, k16, MQ, DIM, DIM);
    gemm_wmma_kernel<0><<<gProj, 256, 0, stream>>>(hN16, wv16 + wOff, nullptr,
                                                   nullptr, v16, MQ, DIM, DIM);
    flash_attn_kernel<<<gAttn, 32, 0, stream>>>(q16, k16, v16, o16);
    gemm_wmma_kernel<2><<<gProj, 256, 0, stream>>>(o16, wo16 + wOff, nullptr,
                                                   carry, nullptr, MQ, DIM, DIM);
    ln_rows_kernel<<<MQ, 256, 0, stream>>>(carry, ln2_g + (long)l * DIM,
                                           ln2_b + (long)l * DIM, hN16);
    gemm_wmma_kernel<1><<<gFF, 256, 0, stream>>>(
        hN16, w116 + (long)l * DIM * FFDIM, mlp_b1 + (long)l * FFDIM, nullptr,
        hid16, MQ, FFDIM, DIM);
    gemm_wmma_kernel<2><<<gProj, 256, 0, stream>>>(
        hid16, w216 + (long)l * FFDIM * DIM, mlp_b2 + (long)l * DIM, carry,
        nullptr, MQ, DIM, FFDIM);
  }

  head_kernel<<<BATCH, 256, 0, stream>>>(carry, lnf_g, lnf_b, head_W, head_b,
                                         (float*)d_out);
}
